// DeformableTransformerDecoderLayer_54546084659226
// MI455X (gfx1250) — compile-verified, matrix operand order checked by reading, not run on hardware
//
#include <hip/hip_runtime.h>

// ---------------------------------------------------------------------------
// Deformable Transformer decoder layer for MI455X (gfx1250, wave32, WMMA)
// All GEMMs + attention use v_wmma_f32_16x16x32_bf16 (f32 accumulate).
// Weights are pre-transposed to bf16 [N][K] so GEMMs need no LDS staging:
// every WMMA fragment is loaded directly with global_load_b128.
// ---------------------------------------------------------------------------

typedef __bf16 bf16;
typedef __attribute__((ext_vector_type(16))) __bf16        v16bf;
typedef __attribute__((ext_vector_type(8)))  float         v8f;
typedef __attribute__((ext_vector_type(4)))  float         float4v;
typedef __attribute__((ext_vector_type(4)))  unsigned int  uint4v;

union fragu { v16bf v; uint4v u[2]; };          // 32B WMMA 16-bit fragment

#define D_MODEL 256
#define N_HEADS 8
#define DH      32
#define LQ      900
#define NB      8
#define LIN     13294   // 100*100 + 50*50 + 25*25 + 13*13
#define D_FFN   1024

// ---------------------------------------------------------------------------
// Weight prep: wt[N][K] (bf16) = transpose(w[K][N] f32). K,N % 32 == 0.
// ---------------------------------------------------------------------------
__global__ __launch_bounds__(256) void transpose_cast(
    const float* __restrict__ w, bf16* __restrict__ wt, int K, int N)
{
  __shared__ float t[32][33];
  const int n0 = blockIdx.x * 32;
  const int k0 = blockIdx.y * 32;
  const int tx = threadIdx.x & 31;
  const int ty = threadIdx.x >> 5;       // 0..7
  #pragma unroll
  for (int i = ty; i < 32; i += 8)
    t[i][tx] = w[(size_t)(k0 + i) * N + n0 + tx];
  __syncthreads();
  #pragma unroll
  for (int i = ty; i < 32; i += 8)
    wt[(size_t)(n0 + i) * K + k0 + tx] = (bf16)t[tx][i];
}

// ---------------------------------------------------------------------------
// GEMM: C[M,N] = A[M,K](f32) * Bt[N,K](bf16, pre-transposed) + bias, opt ReLU.
// K % 32 == 0, N % 128 == 0 (true for 128/256/1024 used here).
// Zero LDS / zero barriers: A fragments are 4x global b128 + cvt_pk,
// B fragments are 2x global b128 per subtile (L2-resident weights).
// 64x128 block tile, 8 waves x four 16x16 WMMA accumulators.
// ---------------------------------------------------------------------------
__global__ __launch_bounds__(256) void gemm_bias_wmma(
    const float* __restrict__ A, const bf16* __restrict__ Bt,
    const float* __restrict__ bias, float* __restrict__ C,
    int M, int N, int K, int relu)
{
  const int tid   = threadIdx.x;
  const int lane  = tid & 31;
  const int wave  = tid >> 5;
  const int m0    = blockIdx.y * 64;
  const int n0    = blockIdx.x * 128;
  const int mi    = wave & 3;                   // 16-row subtile
  const int njb   = (wave >> 2) * 4;            // first of four 16-col subtiles
  const int l15   = lane & 15;
  const int khalf = (lane >> 4) * 8;            // fragment K offset per lane half

  const int  row    = m0 + mi * 16 + l15;
  const bool rvalid = row < M;
  const int  rowc   = rvalid ? row : (M - 1);   // clamp reads, guard stores
  const float* arow = A + (size_t)rowc * K;

  v8f acc[4] = {v8f{}, v8f{}, v8f{}, v8f{}};

  for (int k0 = 0; k0 < K; k0 += 32) {
    // A fragment: elems 0..7 = K[k0+khalf .. +7], 8..15 = K[k0+16+khalf .. +7]
    const float4v a0 = *(const float4v*)(arow + k0 + khalf);
    const float4v a1 = *(const float4v*)(arow + k0 + khalf + 4);
    const float4v a2 = *(const float4v*)(arow + k0 + 16 + khalf);
    const float4v a3 = *(const float4v*)(arow + k0 + 16 + khalf + 4);
    if (k0 + 32 < K) __builtin_prefetch(arow + k0 + 32, 0, 0);  // global_prefetch_b8
    fragu af;
    #pragma unroll
    for (int j = 0; j < 4; ++j) {
      af.v[j]      = (bf16)a0[j];
      af.v[4 + j]  = (bf16)a1[j];
      af.v[8 + j]  = (bf16)a2[j];
      af.v[12 + j] = (bf16)a3[j];
    }
    #pragma unroll
    for (int j = 0; j < 4; ++j) {
      const bf16* bp = Bt + (size_t)(n0 + (njb + j) * 16 + l15) * K + k0 + khalf;
      fragu bfr;
      bfr.u[0] = *(const uint4v*)bp;
      bfr.u[1] = *(const uint4v*)(bp + 16);
      acc[j] = __builtin_amdgcn_wmma_f32_16x16x32_bf16(false, af.v, false, bfr.v,
                                                       (short)0, acc[j], false, false);
    }
  }

  // Epilogue: C/D layout row = r + 8*(lane>=16), col = lane&15.
  const int rbase = m0 + mi * 16 + ((lane >> 4) << 3);
  #pragma unroll
  for (int j = 0; j < 4; ++j) {
    const int col = n0 + (njb + j) * 16 + l15;
    const float bc = bias[col];
    #pragma unroll
    for (int r = 0; r < 8; ++r) {
      const int gr = rbase + r;
      if (gr < M) {
        float v = acc[j][r] + bc;
        if (relu) v = fmaxf(v, 0.0f);
        C[(size_t)gr * N + col] = v;
      }
    }
  }
}

// ---------------------------------------------------------------------------
// Flash-style self-attention. One wave per (batch, head, 16-query block).
// dh = 32 == WMMA K: S(16x32) = Q(16x32) x K^T(32x32) -> 2 WMMAs,
// O += P(16x32) x V(32x32) -> 2 WMMAs per 32-key chunk.
// ---------------------------------------------------------------------------
__global__ __launch_bounds__(32) void flash_attn(
    const float* __restrict__ Q, const float* __restrict__ Kv,
    const float* __restrict__ V, float* __restrict__ O)
{
  __shared__ __align__(16) bf16 P[16][40];  // C-layout -> A-layout bounce

  const int lane  = threadIdx.x;
  const int qb    = blockIdx.x;             // 0..56
  const int h     = blockIdx.y;
  const int n     = blockIdx.z;
  const int l15   = lane & 15;
  const int khalf = (lane >> 4) * 8;
  const float scale = 0.17677669529663687f; // 1/sqrt(32)
  const size_t base = (size_t)n * LQ * D_MODEL + h * DH;

  fragu aq;
  {
    int q = qb * 16 + l15; if (q >= LQ) q = LQ - 1;
    const float* qp = Q + base + (size_t)q * D_MODEL;
    const float4v q0 = *(const float4v*)(qp + khalf);
    const float4v q1 = *(const float4v*)(qp + khalf + 4);
    const float4v q2 = *(const float4v*)(qp + 16 + khalf);
    const float4v q3 = *(const float4v*)(qp + 16 + khalf + 4);
    #pragma unroll
    for (int j = 0; j < 4; ++j) {
      aq.v[j]      = (bf16)(q0[j] * scale);
      aq.v[4 + j]  = (bf16)(q1[j] * scale);
      aq.v[8 + j]  = (bf16)(q2[j] * scale);
      aq.v[12 + j] = (bf16)(q3[j] * scale);
    }
  }

  float mrow[8], lrow[8];
  v8f o0 = {}; v8f o1 = {};
  #pragma unroll
  for (int r = 0; r < 8; ++r) { mrow[r] = -1e30f; lrow[r] = 0.0f; }

  for (int kc = 0; kc < LQ; kc += 32) {
    fragu bk0, bk1;
    {
      int key0 = kc + l15;      if (key0 >= LQ) key0 = LQ - 1;
      int key1 = kc + 16 + l15; if (key1 >= LQ) key1 = LQ - 1;
      const float* kp0 = Kv + base + (size_t)key0 * D_MODEL;
      const float* kp1 = Kv + base + (size_t)key1 * D_MODEL;
      const float4v a0 = *(const float4v*)(kp0 + khalf);
      const float4v a1 = *(const float4v*)(kp0 + khalf + 4);
      const float4v a2 = *(const float4v*)(kp0 + 16 + khalf);
      const float4v a3 = *(const float4v*)(kp0 + 16 + khalf + 4);
      const float4v b0 = *(const float4v*)(kp1 + khalf);
      const float4v b1 = *(const float4v*)(kp1 + khalf + 4);
      const float4v b2 = *(const float4v*)(kp1 + 16 + khalf);
      const float4v b3 = *(const float4v*)(kp1 + 16 + khalf + 4);
      #pragma unroll
      for (int j = 0; j < 4; ++j) {
        bk0.v[j] = (bf16)a0[j]; bk0.v[4+j] = (bf16)a1[j];
        bk0.v[8+j] = (bf16)a2[j]; bk0.v[12+j] = (bf16)a3[j];
        bk1.v[j] = (bf16)b0[j]; bk1.v[4+j] = (bf16)b1[j];
        bk1.v[8+j] = (bf16)b2[j]; bk1.v[12+j] = (bf16)b3[j];
      }
    }
    const v8f z = {};
    v8f s0 = __builtin_amdgcn_wmma_f32_16x16x32_bf16(false, aq.v, false, bk0.v,
                                                     (short)0, z, false, false);
    v8f s1 = __builtin_amdgcn_wmma_f32_16x16x32_bf16(false, aq.v, false, bk1.v,
                                                     (short)0, z, false, false);
    const bool inv0 = (kc + l15)      >= LQ;
    const bool inv1 = (kc + 16 + l15) >= LQ;
    float p0[8], p1[8];
    #pragma unroll
    for (int r = 0; r < 8; ++r) {
      const float a0 = inv0 ? -1e30f : s0[r];
      const float a1 = inv1 ? -1e30f : s1[r];
      float t = fmaxf(a0, a1);
      #pragma unroll
      for (int off = 1; off < 16; off <<= 1)
        t = fmaxf(t, __shfl_xor(t, off, 32));
      const float mn = fmaxf(mrow[r], t);
      const float alpha = __expf(mrow[r] - mn);
      mrow[r] = mn;
      p0[r] = __expf(a0 - mn);
      p1[r] = __expf(a1 - mn);
      float rs = p0[r] + p1[r];
      #pragma unroll
      for (int off = 1; off < 16; off <<= 1)
        rs += __shfl_xor(rs, off, 32);
      lrow[r] = lrow[r] * alpha + rs;
      o0[r] *= alpha;
      o1[r] *= alpha;
    }
    const int prow = (lane >> 4) << 3;
    #pragma unroll
    for (int r = 0; r < 8; ++r) {
      P[prow + r][l15]      = (bf16)p0[r];
      P[prow + r][l15 + 16] = (bf16)p1[r];
    }
    __syncthreads();
    fragu ap;
    ap.u[0] = *(const uint4v*)&P[l15][khalf];
    ap.u[1] = *(const uint4v*)&P[l15][16 + khalf];
    __syncthreads();
    fragu bv0, bv1;
    #pragma unroll
    for (int d = 0; d < 8; ++d) {
      const int kb = ((d & 4) << 2) + khalf + ((d & 3) << 1);
      int k0i = kc + kb;     if (k0i >= LQ) k0i = LQ - 1;
      int k1i = kc + kb + 1; if (k1i >= LQ) k1i = LQ - 1;
      const float* vp0 = V + base + (size_t)k0i * D_MODEL;
      const float* vp1 = V + base + (size_t)k1i * D_MODEL;
      bv0.v[2*d] = (bf16)vp0[l15];       bv0.v[2*d+1] = (bf16)vp1[l15];
      bv1.v[2*d] = (bf16)vp0[l15 + 16];  bv1.v[2*d+1] = (bf16)vp1[l15 + 16];
    }
    o0 = __builtin_amdgcn_wmma_f32_16x16x32_bf16(false, ap.v, false, bv0.v,
                                                 (short)0, o0, false, false);
    o1 = __builtin_amdgcn_wmma_f32_16x16x32_bf16(false, ap.v, false, bv1.v,
                                                 (short)0, o1, false, false);
  }

  #pragma unroll
  for (int r = 0; r < 8; ++r) {
    const int q = qb * 16 + ((lane >> 4) << 3) + r;
    if (q < LQ) {
      const float invl = 1.0f / lrow[r];
      O[base + (size_t)q * D_MODEL + l15]      = o0[r] * invl;
      O[base + (size_t)q * D_MODEL + l15 + 16] = o1[r] * invl;
    }
  }
}

__global__ void ew_add(const float* __restrict__ a, const float* __restrict__ b,
                       float* __restrict__ o, int n)
{
  const int i = blockIdx.x * blockDim.x + threadIdx.x;
  if (i < n) o[i] = a[i] + b[i];
}

__global__ __launch_bounds__(256) void add_ln(
    const float* __restrict__ x, const float* __restrict__ res,
    const float* __restrict__ g, const float* __restrict__ b,
    float* __restrict__ out)
{
  __shared__ float red[16];
  const int row = blockIdx.x;
  const int tid = threadIdx.x;
  const size_t off = (size_t)row * D_MODEL + tid;
  const float v = x[off] + res[off];
  float s = v, sq = v * v;
  #pragma unroll
  for (int o = 1; o < 32; o <<= 1) {
    s  += __shfl_xor(s,  o, 32);
    sq += __shfl_xor(sq, o, 32);
  }
  const int wave = tid >> 5;
  if ((tid & 31) == 0) { red[wave] = s; red[8 + wave] = sq; }
  __syncthreads();
  float ts = 0.0f, tsq = 0.0f;
  #pragma unroll
  for (int i = 0; i < 8; ++i) { ts += red[i]; tsq += red[8 + i]; }
  const float mean = ts * (1.0f / 256.0f);
  const float var  = tsq * (1.0f / 256.0f) - mean * mean;
  const float inv  = rsqrtf(var + 1e-5f);
  out[off] = (v - mean) * inv * g[tid] + b[tid];
}

__global__ void softmax16(float* __restrict__ att, int total)
{
  const int i = blockIdx.x * blockDim.x + threadIdx.x;
  if (i >= total) return;
  const int row = i >> 3, h = i & 7;
  float* p = att + (size_t)row * 128 + h * 16;
  float m = -1e30f;
  #pragma unroll
  for (int j = 0; j < 16; ++j) m = fmaxf(m, p[j]);
  float e[16]; float s = 0.0f;
  #pragma unroll
  for (int j = 0; j < 16; ++j) { e[j] = __expf(p[j] - m); s += e[j]; }
  const float inv = 1.0f / s;
  #pragma unroll
  for (int j = 0; j < 16; ++j) p[j] = e[j] * inv;
}

__global__ __launch_bounds__(256) void msdeform_sample(
    const float* __restrict__ value, const float* __restrict__ ref,
    const float* __restrict__ off,   const float* __restrict__ attw,
    float* __restrict__ out)
{
  const int lvlH[4] = {100, 50, 25, 13};
  const int lvlW[4] = {100, 50, 25, 13};
  const int lvlS[4] = {0, 10000, 12500, 13125};

  const int wave = (blockIdx.x * blockDim.x + threadIdx.x) >> 5;
  const int lane = threadIdx.x & 31;
  const int h = wave & 7;
  const int t = wave >> 3;
  const int q = t % LQ;
  const int n = t / LQ;

  const size_t rowq = (size_t)n * LQ + q;
  const float* offp = off  + rowq * 256 + h * 32;
  const float* attp = attw + rowq * 128 + h * 16;
  float acc = 0.0f;

  #pragma unroll
  for (int l = 0; l < 4; ++l) {
    const int Hl = lvlH[l], Wl = lvlW[l];
    const float rx = ref[rowq * 8 + l * 2 + 0];
    const float ry = ref[rowq * 8 + l * 2 + 1];
    const float* vbase = value + ((size_t)n * LIN + lvlS[l]) * D_MODEL
                               + h * DH + lane;
    #pragma unroll
    for (int p = 0; p < 4; ++p) {
      const float ox = offp[l * 8 + p * 2 + 0];
      const float oy = offp[l * 8 + p * 2 + 1];
      const float x = (rx + ox / (float)Wl) * (float)Wl - 0.5f;
      const float y = (ry + oy / (float)Hl) * (float)Hl - 0.5f;
      const float fx0 = floorf(x), fy0 = floorf(y);
      const int x0 = (int)fx0, y0 = (int)fy0;
      const float fx = x - fx0, fy = y - fy0;
      const float a = attp[l * 4 + p];

      float g00 = 0.f, g10 = 0.f, g01 = 0.f, g11 = 0.f;
      const bool vx0 = (x0 >= 0) && (x0 < Wl);
      const bool vx1 = (x0 + 1 >= 0) && (x0 + 1 < Wl);
      const bool vy0 = (y0 >= 0) && (y0 < Hl);
      const bool vy1 = (y0 + 1 >= 0) && (y0 + 1 < Hl);
      if (vx0 && vy0) g00 = vbase[(size_t)(y0 * Wl + x0) * D_MODEL];
      if (vx1 && vy0) g10 = vbase[(size_t)(y0 * Wl + x0 + 1) * D_MODEL];
      if (vx0 && vy1) g01 = vbase[(size_t)((y0 + 1) * Wl + x0) * D_MODEL];
      if (vx1 && vy1) g11 = vbase[(size_t)((y0 + 1) * Wl + x0 + 1) * D_MODEL];

      acc += a * (g00 * (1.f - fx) * (1.f - fy) + g10 * fx * (1.f - fy)
                + g01 * (1.f - fx) * fy         + g11 * fx * fy);
    }
  }
  out[rowq * 256 + h * 32 + lane] = acc;
}

extern "C" void kernel_launch(void* const* d_in, const int* in_sizes, int n_in,
                              void* d_out, int out_size, void* d_ws, size_t ws_size,
                              hipStream_t stream)
{
  const float* tgt   = (const float*)d_in[1];
  const float* qpos  = (const float*)d_in[2];
  const float* ref   = (const float*)d_in[3];
  const float* src   = (const float*)d_in[4];
  const float* wq = (const float*)d_in[8];   const float* bq = (const float*)d_in[9];
  const float* wk = (const float*)d_in[10];  const float* bk = (const float*)d_in[11];
  const float* wv = (const float*)d_in[12];  const float* bv = (const float*)d_in[13];
  const float* wo = (const float*)d_in[14];  const float* bo = (const float*)d_in[15];
  const float* w_val = (const float*)d_in[16]; const float* b_val = (const float*)d_in[17];
  const float* w_off = (const float*)d_in[18]; const float* b_off = (const float*)d_in[19];
  const float* w_att = (const float*)d_in[20]; const float* b_att = (const float*)d_in[21];
  const float* w_out = (const float*)d_in[22]; const float* b_out = (const float*)d_in[23];
  const float* ln1_g = (const float*)d_in[24]; const float* ln1_b = (const float*)d_in[25];
  const float* ln2_g = (const float*)d_in[26]; const float* ln2_b = (const float*)d_in[27];
  const float* ln3_g = (const float*)d_in[28]; const float* ln3_b = (const float*)d_in[29];
  const float* w1 = (const float*)d_in[30];  const float* b1 = (const float*)d_in[31];
  const float* w2 = (const float*)d_in[32];  const float* b2 = (const float*)d_in[33];

  const int M  = NB * LQ;
  const size_t E = (size_t)M * D_MODEL;

  float* ws    = (float*)d_ws;
  float* value = ws;                                   // 27,226,112
  float* big1  = value + (size_t)NB * LIN * D_MODEL;
  float* big2  = big1 + E;
  float* big3  = big2 + E;
  float* big4  = big3 + E;
  float* tA    = big4 + E;
  float* attl  = tA + E;                               // 921,600
  float* tB    = attl + (size_t)M * 128;
  float* hbuf  = tB + E;                               // 7,372,800

  bf16* wq_t   = (bf16*)(hbuf + (size_t)M * D_FFN);    // [256][256]
  bf16* wk_t   = wq_t   + 65536;
  bf16* wv_t   = wk_t   + 65536;
  bf16* wo_t   = wv_t   + 65536;
  bf16* wval_t = wo_t   + 65536;
  bf16* woff_t = wval_t + 65536;
  bf16* wout_t = woff_t + 65536;
  bf16* watt_t = wout_t + 65536;        // [128][256]
  bf16* w1_t   = watt_t + 32768;        // [1024][256]
  bf16* w2_t   = w1_t   + 262144;       // [256][1024]

  const dim3 blk256(256);
  const dim3 tg256(8, 8);
  const dim3 gemm_g(2, (M + 63) / 64);

  transpose_cast<<<tg256, blk256, 0, stream>>>(wq,    wq_t,   256, 256);
  transpose_cast<<<tg256, blk256, 0, stream>>>(wk,    wk_t,   256, 256);
  transpose_cast<<<tg256, blk256, 0, stream>>>(wv,    wv_t,   256, 256);
  transpose_cast<<<tg256, blk256, 0, stream>>>(wo,    wo_t,   256, 256);
  transpose_cast<<<tg256, blk256, 0, stream>>>(w_val, wval_t, 256, 256);
  transpose_cast<<<tg256, blk256, 0, stream>>>(w_off, woff_t, 256, 256);
  transpose_cast<<<tg256, blk256, 0, stream>>>(w_out, wout_t, 256, 256);
  transpose_cast<<<dim3(4, 8),  blk256, 0, stream>>>(w_att, watt_t, 256, 128);
  transpose_cast<<<dim3(32, 8), blk256, 0, stream>>>(w1,    w1_t,   256, 1024);
  transpose_cast<<<dim3(8, 32), blk256, 0, stream>>>(w2,    w2_t,   1024, 256);

  // ---- self attention ----
  ew_add<<<(int)((E + 255) / 256), blk256, 0, stream>>>(tgt, qpos, big1, (int)E);
  gemm_bias_wmma<<<gemm_g, blk256, 0, stream>>>(big1, wq_t, bq, big2, M, 256, 256, 0);
  gemm_bias_wmma<<<gemm_g, blk256, 0, stream>>>(big1, wk_t, bk, big3, M, 256, 256, 0);
  gemm_bias_wmma<<<gemm_g, blk256, 0, stream>>>(tgt,  wv_t, bv, big4, M, 256, 256, 0);
  flash_attn<<<dim3((LQ + 15) / 16, N_HEADS, NB), dim3(32), 0, stream>>>(big2, big3, big4, big1);
  gemm_bias_wmma<<<gemm_g, blk256, 0, stream>>>(big1, wo_t, bo, big3, M, 256, 256, 0);
  add_ln<<<M, blk256, 0, stream>>>(big3, tgt, ln2_g, ln2_b, tA);

  // ---- MS deformable cross attention ----
  gemm_bias_wmma<<<dim3(2, (NB * LIN + 63) / 64), blk256, 0, stream>>>(
      src, wval_t, b_val, value, NB * LIN, 256, 256, 0);
  ew_add<<<(int)((E + 255) / 256), blk256, 0, stream>>>(tA, qpos, big1, (int)E);
  gemm_bias_wmma<<<gemm_g, blk256, 0, stream>>>(big1, woff_t, b_off, big2, M, 256, 256, 0);
  gemm_bias_wmma<<<dim3(1, (M + 63) / 64), blk256, 0, stream>>>(
      big1, watt_t, b_att, attl, M, 128, 256, 0);
  softmax16<<<(M * N_HEADS + 255) / 256, blk256, 0, stream>>>(attl, M * N_HEADS);
  msdeform_sample<<<M * N_HEADS / 8, blk256, 0, stream>>>(value, ref, big2, attl, big4);
  gemm_bias_wmma<<<gemm_g, blk256, 0, stream>>>(big4, wout_t, b_out, big3, M, 256, 256, 0);
  add_ln<<<M, blk256, 0, stream>>>(big3, tA, ln1_g, ln1_b, tB);

  // ---- FFN ----
  gemm_bias_wmma<<<dim3(8, (M + 63) / 64), blk256, 0, stream>>>(
      tB, w1_t, b1, hbuf, M, D_FFN, 256, 1);
  gemm_bias_wmma<<<gemm_g, blk256, 0, stream>>>(hbuf, w2_t, b2, big1, M, 256, D_FFN, 0);
  add_ln<<<M, blk256, 0, stream>>>(big1, tB, ln3_g, ln3_b, (float*)d_out);
}
